// MLPMessagePassingLayer_21406117003777
// MI455X (gfx1250) — compile-verified, matrix operand order checked by reading, not run on hardware
//
#include <hip/hip_runtime.h>
#include <stdint.h>

// Problem sizes (fixed by the reference)
#define NN    100000
#define DD    64
#define EE    1600000
#define MSGH  128
#define MSGO  64
#define NODEH 128

typedef _Float16 v4h  __attribute__((ext_vector_type(4)));
typedef _Float16 v8h  __attribute__((ext_vector_type(8)));
typedef _Float16 v16h __attribute__((ext_vector_type(16)));
typedef float    v8f  __attribute__((ext_vector_type(8)));

// ---------------------------------------------------------------------------
// Load one 16x32 f16 WMMA fragment from a row-major [rows][ldk] f16 LDS array.
// CDNA5 16-bit A/B layout: lane<16 -> row=lane,  K = {k0..k0+7, k0+16..k0+23}
//                          lane>=16 -> row=lane-16, K shifted by +8.
// Both A (M-major) and B (N-major, i.e. W^T) use the same striping.
// ---------------------------------------------------------------------------
__device__ __forceinline__ v16h ld_frag(const _Float16* p0, int ldk, int row0,
                                        int k0, int lane) {
  const int r  = lane & 15;
  const int hi = lane >> 4;
  const _Float16* p = p0 + (size_t)(row0 + r) * ldk + k0 + hi * 8;
  v8h lo = *(const v8h*)p;
  v8h hs = *(const v8h*)(p + 16);
  v16h out;
#pragma unroll
  for (int i = 0; i < 8; ++i) { out[i] = lo[i]; out[i + 8] = hs[i]; }
  return out;
}

// Native fp32 global atomic add (fire-and-forget; S_ENDPGM drains STOREcnt).
__device__ __forceinline__ void atomic_add_f32(float* p, float v) {
  asm volatile("global_atomic_add_f32 %0, %1, off"
               :: "v"((uint64_t)(uintptr_t)p), "v"(v) : "memory");
}

// ---------------------------------------------------------------------------
// Kernel 0: zero agg + convert/transpose weights fp32 -> f16 [N][K] layout
// ---------------------------------------------------------------------------
__global__ __launch_bounds__(256) void prep_kernel(
    const float* __restrict__ mW1, const float* __restrict__ mW2,
    const float* __restrict__ nW1, const float* __restrict__ nW2,
    float* __restrict__ agg,
    _Float16* __restrict__ W1T, _Float16* __restrict__ W2T,
    _Float16* __restrict__ NW1T, _Float16* __restrict__ NW2T) {
  const int t  = blockIdx.x * 256 + threadIdx.x;
  const int nt = gridDim.x * 256;
  for (int i = t; i < NN * MSGO; i += nt) agg[i] = 0.f;
  for (int i = t; i < MSGH * 192; i += nt) {           // msg_W1 [192][128] -> [128][192]
    int n = i / 192, k = i % 192;
    W1T[i] = (_Float16)mW1[k * MSGH + n];
  }
  for (int i = t; i < MSGO * MSGH; i += nt) {          // msg_W2 [128][64] -> [64][128]
    int n = i / MSGH, k = i % MSGH;
    W2T[i] = (_Float16)mW2[k * MSGO + n];
  }
  for (int i = t; i < NODEH * 128; i += nt) {          // node_W1 [128][128] -> [128][128]
    int n = i / 128, k = i % 128;
    NW1T[i] = (_Float16)nW1[k * NODEH + n];
  }
  for (int i = t; i < DD * NODEH; i += nt) {           // node_W2 [128][64] -> [64][128]
    int n = i / NODEH, k = i % NODEH;
    NW2T[i] = (_Float16)nW2[k * DD + n];
  }
}

// ---------------------------------------------------------------------------
// Kernel 1: message MLP + atomic aggregation.  128 edges / block, 8 waves.
// ---------------------------------------------------------------------------
__global__ __launch_bounds__(256) void msg_kernel(
    const float* __restrict__ nodes, const float* __restrict__ edges,
    const int* __restrict__ senders, const int* __restrict__ receivers,
    const float* __restrict__ b1, const float* __restrict__ b2,
    const _Float16* __restrict__ W1T, const _Float16* __restrict__ W2T,
    float* __restrict__ agg) {
  // Padded strides: 200 halves (100 dw, 36r%64 bank walk) / 136 halves (68 dw)
  __shared__ _Float16 Xs[128][200];   // concat(sender,recv,edge) tile, K=192
  __shared__ _Float16 W1s[128][200];  // W1^T  [N=128][K=192]
  __shared__ _Float16 W2s[64][136];   // W2^T  [N=64][K=128]
  __shared__ _Float16 Hs[128][136];   // hidden [128][128]
  __shared__ int sIdx[128];
  __shared__ int rIdx[128];

  const int tid = threadIdx.x;
  const int e0  = blockIdx.x * 128;

  // Stage f16 weights (L2-resident) into LDS.
  for (int i = tid; i < (MSGH * 192) / 8; i += 256) {
    int n = i / 24, c = (i % 24) * 8;
    *(v8h*)&W1s[n][c] = *(const v8h*)(W1T + n * 192 + c);
  }
  for (int i = tid; i < (MSGO * MSGH) / 8; i += 256) {
    int n = i / 16, c = (i % 16) * 8;
    *(v8h*)&W2s[n][c] = *(const v8h*)(W2T + n * MSGH + c);
  }
  if (tid < 128) {
    sIdx[tid] = senders[e0 + tid];
    rIdx[tid] = receivers[e0 + tid];
  }
  __syncthreads();

  // Gather + fp32->f16 convert into X tile: [s | r | e] = 192 cols.
  for (int i = tid; i < 128 * 16; i += 256) {
    int row = i >> 4, q = (i & 15) << 2;
    float4 a = *(const float4*)(nodes + (size_t)sIdx[row] * DD + q);
    float4 b = *(const float4*)(nodes + (size_t)rIdx[row] * DD + q);
    float4 e = *(const float4*)(edges + (size_t)(e0 + row) * DD + q);
    v4h ha = {(_Float16)a.x, (_Float16)a.y, (_Float16)a.z, (_Float16)a.w};
    v4h hb = {(_Float16)b.x, (_Float16)b.y, (_Float16)b.z, (_Float16)b.w};
    v4h he = {(_Float16)e.x, (_Float16)e.y, (_Float16)e.z, (_Float16)e.w};
    *(v4h*)&Xs[row][q]       = ha;
    *(v4h*)&Xs[row][64 + q]  = hb;
    *(v4h*)&Xs[row][128 + q] = he;
  }
  __syncthreads();

  const int wave = tid >> 5, lane = tid & 31;
  const int m0 = wave * 16;
  const int r15 = lane & 15, hi = lane >> 4;

  // ---- GEMM1: X[128x192] @ W1[192x128] (+b1, ReLU) ----
  v16h a1[6];
#pragma unroll
  for (int kk = 0; kk < 6; ++kk) a1[kk] = ld_frag(&Xs[0][0], 200, m0, kk * 32, lane);
#pragma unroll
  for (int n = 0; n < 8; ++n) {
    float bv = b1[n * 16 + r15];
    v8f c = {bv, bv, bv, bv, bv, bv, bv, bv};
#pragma unroll
    for (int kk = 0; kk < 6; ++kk) {
      v16h b = ld_frag(&W1s[0][0], 200, n * 16, kk * 32, lane);
      c = __builtin_amdgcn_wmma_f32_16x16x32_f16(false, a1[kk], false, b,
                                                 (short)0, c, false, false);
    }
#pragma unroll
    for (int i = 0; i < 8; ++i) {
      float v = c[i] > 0.f ? c[i] : 0.f;
      Hs[m0 + hi * 8 + i][n * 16 + r15] = (_Float16)v;  // wave-private rows
    }
  }

  // ---- GEMM2: H[128x128] @ W2[128x64] (+b2, ReLU) -> atomic scatter ----
  v16h a2[4];
#pragma unroll
  for (int kk = 0; kk < 4; ++kk) a2[kk] = ld_frag(&Hs[0][0], 136, m0, kk * 32, lane);
#pragma unroll
  for (int n = 0; n < 4; ++n) {
    float bv = b2[n * 16 + r15];
    v8f c = {bv, bv, bv, bv, bv, bv, bv, bv};
#pragma unroll
    for (int kk = 0; kk < 4; ++kk) {
      v16h b = ld_frag(&W2s[0][0], 136, n * 16, kk * 32, lane);
      c = __builtin_amdgcn_wmma_f32_16x16x32_f16(false, a2[kk], false, b,
                                                 (short)0, c, false, false);
    }
    const int col = n * 16 + r15;
#pragma unroll
    for (int i = 0; i < 8; ++i) {
      int row = m0 + hi * 8 + i;
      float v = c[i] > 0.f ? c[i] : 0.f;
      atomic_add_f32(agg + (size_t)rIdx[row] * MSGO + col, v);
    }
  }
}

// ---------------------------------------------------------------------------
// Kernel 2: node MLP + LayerNorm + residual.  128 nodes / block, 8 waves.
// ---------------------------------------------------------------------------
__global__ __launch_bounds__(256) void node_kernel(
    const float* __restrict__ nodes, const float* __restrict__ agg,
    const float* __restrict__ b1, const float* __restrict__ b2,
    const float* __restrict__ ln_scale, const float* __restrict__ ln_offset,
    const _Float16* __restrict__ NW1T, const _Float16* __restrict__ NW2T,
    float* __restrict__ out) {
  __shared__ _Float16 Ys[128][136];   // concat(nodes, agg), K=128
  __shared__ _Float16 W1s[128][136];  // node_W1^T
  __shared__ _Float16 W2s[64][136];   // node_W2^T
  __shared__ _Float16 Hs[128][136];
  __shared__ float    Yo[128][68];    // fp32 pre-LN output

  const int tid = threadIdx.x;
  const int n0  = blockIdx.x * 128;

  for (int i = tid; i < (NODEH * 128) / 8; i += 256) {
    int n = i >> 4, c = (i & 15) * 8;
    *(v8h*)&W1s[n][c] = *(const v8h*)(NW1T + n * 128 + c);
  }
  for (int i = tid; i < (DD * NODEH) / 8; i += 256) {
    int n = i >> 4, c = (i & 15) * 8;
    *(v8h*)&W2s[n][c] = *(const v8h*)(NW2T + n * NODEH + c);
  }
  for (int i = tid; i < 128 * 16; i += 256) {
    int row = i >> 4, q = (i & 15) << 2;
    int g = n0 + row; if (g >= NN) g = NN - 1;           // clamp pad rows
    float4 a = *(const float4*)(nodes + (size_t)g * DD + q);
    float4 b = *(const float4*)(agg + (size_t)g * MSGO + q);
    v4h ha = {(_Float16)a.x, (_Float16)a.y, (_Float16)a.z, (_Float16)a.w};
    v4h hb = {(_Float16)b.x, (_Float16)b.y, (_Float16)b.z, (_Float16)b.w};
    *(v4h*)&Ys[row][q]      = ha;
    *(v4h*)&Ys[row][64 + q] = hb;
  }
  __syncthreads();

  const int wave = tid >> 5, lane = tid & 31;
  const int m0 = wave * 16;
  const int r15 = lane & 15, hi = lane >> 4;

  // ---- GEMM1: Y[128x128] @ node_W1[128x128] (+b1, ReLU) ----
  v16h a1[4];
#pragma unroll
  for (int kk = 0; kk < 4; ++kk) a1[kk] = ld_frag(&Ys[0][0], 136, m0, kk * 32, lane);
#pragma unroll
  for (int n = 0; n < 8; ++n) {
    float bv = b1[n * 16 + r15];
    v8f c = {bv, bv, bv, bv, bv, bv, bv, bv};
#pragma unroll
    for (int kk = 0; kk < 4; ++kk) {
      v16h b = ld_frag(&W1s[0][0], 136, n * 16, kk * 32, lane);
      c = __builtin_amdgcn_wmma_f32_16x16x32_f16(false, a1[kk], false, b,
                                                 (short)0, c, false, false);
    }
#pragma unroll
    for (int i = 0; i < 8; ++i) {
      float v = c[i] > 0.f ? c[i] : 0.f;
      Hs[m0 + hi * 8 + i][n * 16 + r15] = (_Float16)v;
    }
  }

  // ---- GEMM2: H[128x128] @ node_W2[128x64] (+b2, no activation) ----
  v16h a2[4];
#pragma unroll
  for (int kk = 0; kk < 4; ++kk) a2[kk] = ld_frag(&Hs[0][0], 136, m0, kk * 32, lane);
#pragma unroll
  for (int n = 0; n < 4; ++n) {
    float bv = b2[n * 16 + r15];
    v8f c = {bv, bv, bv, bv, bv, bv, bv, bv};
#pragma unroll
    for (int kk = 0; kk < 4; ++kk) {
      v16h b = ld_frag(&W2s[0][0], 136, n * 16, kk * 32, lane);
      c = __builtin_amdgcn_wmma_f32_16x16x32_f16(false, a2[kk], false, b,
                                                 (short)0, c, false, false);
    }
#pragma unroll
    for (int i = 0; i < 8; ++i) Yo[m0 + hi * 8 + i][n * 16 + r15] = c[i];
  }
  __syncthreads();

  // ---- LayerNorm + residual: one thread per node row ----
  if (tid < 128) {
    int g = n0 + tid;
    if (g < NN) {
      float mean = 0.f;
#pragma unroll
      for (int c = 0; c < 64; ++c) mean += Yo[tid][c];
      mean *= (1.f / 64.f);
      float var = 0.f;
#pragma unroll
      for (int c = 0; c < 64; ++c) {
        float d = Yo[tid][c] - mean;
        var += d * d;
      }
      var *= (1.f / 64.f);
      float inv = rsqrtf(var + 1e-5f);
#pragma unroll
      for (int c = 0; c < 64; ++c) {
        float y = (Yo[tid][c] - mean) * inv * ln_scale[c] + ln_offset[c];
        out[(size_t)g * DD + c] = nodes[(size_t)g * DD + c] + y;
      }
    }
  }
}

// ---------------------------------------------------------------------------
// Host launcher.  Workspace: agg (25.6 MB) + f16 weights (~115 KB).
// ---------------------------------------------------------------------------
extern "C" void kernel_launch(void* const* d_in, const int* in_sizes, int n_in,
                              void* d_out, int out_size, void* d_ws,
                              size_t ws_size, hipStream_t stream) {
  const float* nodes     = (const float*)d_in[0];
  const float* edges     = (const float*)d_in[1];
  const int*   senders   = (const int*)d_in[2];
  const int*   receivers = (const int*)d_in[3];
  const float* msg_W1    = (const float*)d_in[4];
  const float* msg_b1    = (const float*)d_in[5];
  const float* msg_W2    = (const float*)d_in[6];
  const float* msg_b2    = (const float*)d_in[7];
  const float* node_W1   = (const float*)d_in[8];
  const float* node_b1   = (const float*)d_in[9];
  const float* node_W2   = (const float*)d_in[10];
  const float* node_b2   = (const float*)d_in[11];
  const float* ln_scale  = (const float*)d_in[12];
  const float* ln_offset = (const float*)d_in[13];
  float* out = (float*)d_out;

  char* ws = (char*)d_ws;
  const size_t AGG_BYTES = (size_t)NN * MSGO * sizeof(float);  // 25,600,000 (256-aligned)
  float*    agg  = (float*)ws;
  _Float16* W1T  = (_Float16*)(ws + AGG_BYTES);
  _Float16* W2T  = W1T + MSGH * 192;
  _Float16* NW1T = W2T + MSGO * MSGH;
  _Float16* NW2T = NW1T + NODEH * 128;

  prep_kernel<<<512, 256, 0, stream>>>(msg_W1, msg_W2, node_W1, node_W2, agg,
                                       W1T, W2T, NW1T, NW2T);
  msg_kernel<<<EE / 128, 256, 0, stream>>>(nodes, edges, senders, receivers,
                                           msg_b1, msg_b2, W1T, W2T, agg);
  node_kernel<<<(NN + 127) / 128, 256, 0, stream>>>(nodes, agg, node_b1,
                                                    node_b2, ln_scale,
                                                    ln_offset, NW1T, NW2T, out);
}
